// MoELayer_18571438588048
// MI455X (gfx1250) — matrix operand order repaired and checked by reading, not compiled, not run on hardware
//
#include <hip/hip_runtime.h>
#include <hip/hip_bf16.h>
#include <math.h>

// Problem constants
#define BB 4096
#define DD 1024
#define OO 1024
#define EE 8
#define HH 4096
#define KK 2

typedef __attribute__((ext_vector_type(16))) __bf16 v16bf;
typedef __attribute__((ext_vector_type(8)))  __bf16 v8bf;
typedef __attribute__((ext_vector_type(4)))  __bf16 v4bf;
typedef __attribute__((ext_vector_type(8)))  float  v8f;

#define TM 64     // rows per block tile
#define TN 128    // cols per block tile
#define TK 32     // K-step (one WMMA depth)
#define BPAD 40   // padded LDS row stride (bf16 elems): 80B keeps 16B alignment

// ---------------------------------------------------------------------------
// Kernel 0: zero counts + fill counters (16 ints)
// ---------------------------------------------------------------------------
__global__ void moe_zero16(int* p) {
    if (threadIdx.x < 16) p[threadIdx.x] = 0;
}

// ---------------------------------------------------------------------------
// Kernel 1: router — one wave (32 lanes) per token
// ---------------------------------------------------------------------------
__global__ __launch_bounds__(256) void moe_router(
    const float* __restrict__ x, const float* __restrict__ gw,
    const float* __restrict__ gb, int* __restrict__ counts,
    int2* __restrict__ tkidx, float2* __restrict__ tkw) {
    int wave = threadIdx.x >> 5, lane = threadIdx.x & 31;
    int b = blockIdx.x * 8 + wave;
    float acc[EE];
#pragma unroll
    for (int e = 0; e < EE; ++e) acc[e] = 0.f;
    const float* xr = x + (size_t)b * DD;
    for (int k = lane; k < DD; k += 32) {
        float xv = xr[k];
        const float* g = gw + (size_t)k * EE;
#pragma unroll
        for (int e = 0; e < EE; ++e) acc[e] += xv * g[e];
    }
#pragma unroll
    for (int e = 0; e < EE; ++e)
        for (int off = 16; off > 0; off >>= 1)
            acc[e] += __shfl_down(acc[e], off, 32);
    if (lane == 0) {
        float v[EE];
#pragma unroll
        for (int e = 0; e < EE; ++e) v[e] = acc[e] + gb[e];
        int i0 = 0; float v0 = v[0];
#pragma unroll
        for (int e = 1; e < EE; ++e) if (v[e] > v0) { v0 = v[e]; i0 = e; }
        int i1 = (i0 == 0) ? 1 : 0; float v1 = v[i1];
#pragma unroll
        for (int e = 0; e < EE; ++e)
            if (e != i0 && v[e] > v1) { v1 = v[e]; i1 = e; }
        float w0 = 1.f / (1.f + __expf(v1 - v0));   // softmax over {v0,v1}
        tkidx[b] = make_int2(i0, i1);
        tkw[b]   = make_float2(w0, 1.f - w0);
        atomicAdd(&counts[i0], 1);
        atomicAdd(&counts[i1], 1);
    }
}

// ---------------------------------------------------------------------------
// Kernel 2: exclusive scan over 8 expert counts
// ---------------------------------------------------------------------------
__global__ void moe_scan(const int* __restrict__ counts, int* __restrict__ offs) {
    if (threadIdx.x == 0) {
        int run = 0;
#pragma unroll
        for (int e = 0; e < EE; ++e) { offs[e] = run; run += counts[e]; }
    }
}

// ---------------------------------------------------------------------------
// Kernel 3: scatter tokens into expert-grouped row lists
// ---------------------------------------------------------------------------
__global__ __launch_bounds__(256) void moe_scatter(
    const int2* __restrict__ tkidx, const float2* __restrict__ tkw,
    const int* __restrict__ offs, int* __restrict__ fill,
    int* __restrict__ rtok, int* __restrict__ rdst, float* __restrict__ rwt) {
    int b = blockIdx.x * 256 + threadIdx.x;
    if (b >= BB) return;
    int2 id = tkidx[b]; float2 w = tkw[b];
    int p0 = offs[id.x] + atomicAdd(&fill[id.x], 1);
    rtok[p0] = b; rdst[p0] = b * KK + 0; rwt[p0] = w.x;
    int p1 = offs[id.y] + atomicAdd(&fill[id.y], 1);
    rtok[p1] = b; rdst[p1] = b * KK + 1; rwt[p1] = w.y;
}

// ---------------------------------------------------------------------------
// Per-lane WMMA fragment gathers (ISA 7.12.2 16-bit layouts, wave32)
// ---------------------------------------------------------------------------
__device__ inline v16bf frag_a(const __bf16* Ast, int lane) {
    int row = lane & 15, half = lane >> 4;            // lanes 16-31: K+8 / K+24
    const v8bf* lo = (const v8bf*)(Ast + row * BPAD + half * 8);
    const v8bf* hi = (const v8bf*)(Ast + row * BPAD + half * 8 + 16);
    return __builtin_shufflevector(*lo, *hi, 0,1,2,3,4,5,6,7,8,9,10,11,12,13,14,15);
}
__device__ inline v16bf frag_b(const __bf16* Bst, int lane, int n_off) {
    int n = n_off + (lane & 15), k0 = (lane >> 4) * 8;  // lanes 16-31: K+8 / K+24
    const v8bf* lo = (const v8bf*)(Bst + n * BPAD + k0);
    const v8bf* hi = (const v8bf*)(Bst + n * BPAD + k0 + 16);
    return __builtin_shufflevector(*lo, *hi, 0,1,2,3,4,5,6,7,8,9,10,11,12,13,14,15);
}

__device__ inline void cvt_store4(__bf16* p, float4 v) {
    v4bf o;
    o.x = (__bf16)v.x; o.y = (__bf16)v.y; o.z = (__bf16)v.z; o.w = (__bf16)v.w;
    *(v4bf*)p = o;     // single ds_store_b64
}

// async global->LDS copy of 8 bytes (ASYNCcnt-tracked, gfx1250)
__device__ inline void async_copy_b64(unsigned lds_off, const void* gaddr) {
    asm volatile("global_load_async_to_lds_b64 %0, %1, off"
                 :: "v"(lds_off), "v"(gaddr) : "memory");
}
__device__ inline void wait_async0() {
    asm volatile("s_wait_asynccnt 0x0" ::: "memory");
}

// ---------------------------------------------------------------------------
// Kernel 4: grouped GEMM #1  h = relu(X_sel @ w1[e] + b1[e])   [rows x HH] bf16
//   block = 256 threads = 8 waves -> 64x128 tile; wave w: 16-col strip,
//   4 M-subtiles -> 4 WMMAs/K-step sharing one B fragment.
// ---------------------------------------------------------------------------
__global__ __launch_bounds__(256) void moe_h_gemm(
    const float* __restrict__ x, const float* __restrict__ w1,
    const float* __restrict__ b1, const int* __restrict__ counts,
    const int* __restrict__ offs, const int* __restrict__ rtok,
    __bf16* __restrict__ h) {
    int e = blockIdx.z, mt = blockIdx.y, nt = blockIdx.x;
    int cnt = counts[e];
    int m0 = mt * TM;
    if (m0 >= cnt) return;
    int base = offs[e];
    int valid = min(TM, cnt - m0);

    __shared__ __bf16 Ast[TM * BPAD];
    __shared__ __bf16 Bst[TN * BPAD];
    __shared__ __bf16 Cst[TM][TN];
    __shared__ int    tok[TM];

    int t = threadIdx.x, lane = t & 31, wave = t >> 5;
    if (t < TM) tok[t] = rtok[base + m0 + min(t, valid - 1)];
    __syncthreads();

    // hoisted A-row pointers (each thread stages rows t>>3 and (t>>3)+32)
    int mA = t >> 3, k4 = (t & 7) * 4;
    const float* xr0 = x + (size_t)tok[mA] * DD + k4;
    const float* xr1 = x + (size_t)tok[mA + 32] * DD + k4;

    int n0 = nt * TN;
    v8f acc[4];
#pragma unroll
    for (int s = 0; s < 4; ++s) acc[s] = (v8f){};

    for (int kk = 0; kk < DD; kk += TK) {
        // stage A tile 64x32: float4 loads, packed bf16x4 LDS stores
        cvt_store4(&Ast[mA * BPAD + k4],        *(const float4*)(xr0 + kk));
        cvt_store4(&Ast[(mA + 32) * BPAD + k4], *(const float4*)(xr1 + kk));
        // stage B tile 32x128 transposed -> [n][k] (f32 -> bf16)
        const float* wsrc = w1 + ((size_t)e * DD + kk) * HH + n0;
#pragma unroll
        for (int i = 0; i < 16; ++i) {
            int idx = i * 256 + t;
            int k = idx >> 7, n = idx & 127;
            Bst[n * BPAD + k] = (__bf16)wsrc[(size_t)k * HH + n];
        }
        if (kk + TK < DD)  // prefetch next K-block of w1 (global_prefetch_b8)
            __builtin_prefetch(wsrc + (size_t)TK * HH + (t >> 3) * HH + (t & 7) * 16, 0, 3);
        __syncthreads();
        v16bf bf = frag_b(Bst, lane, wave * 16);
#pragma unroll
        for (int s = 0; s < 4; ++s) {
            v16bf af = frag_a(Ast + s * 16 * BPAD, lane);
            acc[s] = __builtin_amdgcn_wmma_f32_16x16x32_bf16(false, af, false, bf,
                                                             (short)0, acc[s], false, false);
        }
        __syncthreads();
    }
    // epilogue: bias + relu -> LDS -> coalesced packed stores
    int cn = wave * 16 + (lane & 15);
    float bias = b1[(size_t)e * HH + n0 + cn];
#pragma unroll
    for (int s = 0; s < 4; ++s)
#pragma unroll
        for (int r = 0; r < 8; ++r) {
            int m = s * 16 + r + 8 * (lane >> 4);
            Cst[m][cn] = (__bf16)fmaxf(acc[s][r] + bias, 0.f);
        }
    __syncthreads();
#pragma unroll
    for (int i = 0; i < 8; ++i) {
        int idx = i * 256 + t;
        int m = idx >> 5, n4 = (idx & 31) * 4;
        if (m < valid)
            *(v4bf*)&h[(size_t)(base + m0 + m) * HH + n0 + n4] =
                *(const v4bf*)&Cst[m][n4];
    }
}

// ---------------------------------------------------------------------------
// Kernel 5: grouped GEMM #2  y_slot = (h @ w2[e] + b2[e]) * routing_w
//   A tile is raw bf16 -> staged with async global->LDS copies
// ---------------------------------------------------------------------------
__global__ __launch_bounds__(256) void moe_y_gemm(
    const __bf16* __restrict__ h, const float* __restrict__ w2,
    const float* __restrict__ b2, const int* __restrict__ counts,
    const int* __restrict__ offs, const int* __restrict__ rdst,
    const float* __restrict__ rwt, float* __restrict__ yslot) {
    int e = blockIdx.z, mt = blockIdx.y, nt = blockIdx.x;
    int cnt = counts[e];
    int m0 = mt * TM;
    if (m0 >= cnt) return;
    int base = offs[e];
    int valid = min(TM, cnt - m0);

    __shared__ __bf16 Ast[TM * BPAD];
    __shared__ __bf16 Bst[TN * BPAD];
    __shared__ float  Yst[TM][TN];
    __shared__ int    dstl[TM];
    __shared__ float  wtl[TM];

    int t = threadIdx.x, lane = t & 31, wave = t >> 5;
    if (t < TM) {
        int r = base + m0 + min(t, valid - 1);
        dstl[t] = rdst[r];
        wtl[t]  = rwt[r];
    }

    // hoisted A-row pointers (rows may over-read into padded scratch slack)
    int mA = t >> 3, k4 = (t & 7) * 4;
    const __bf16* hr0 = h + (size_t)(base + m0 + mA) * HH + k4;
    const __bf16* hr1 = h + (size_t)(base + m0 + mA + 32) * HH + k4;
    unsigned abase = (unsigned)(size_t)(&Ast[0]);           // LDS byte offset
    unsigned ad0 = abase + (unsigned)(mA * BPAD + k4) * 2u;
    unsigned ad1 = abase + (unsigned)((mA + 32) * BPAD + k4) * 2u;
    __syncthreads();

    int n0 = nt * TN;
    v8f acc[4];
#pragma unroll
    for (int s = 0; s < 4; ++s) acc[s] = (v8f){};

    for (int kk = 0; kk < HH; kk += TK) {
        // stage A tile 64x32 bf16 via async global->LDS (no VGPR round trip)
        async_copy_b64(ad0, hr0 + kk);
        async_copy_b64(ad1, hr1 + kk);
        // stage B tile 32x128 transposed -> [n][k] (f32 -> bf16)
        const float* wsrc = w2 + ((size_t)e * HH + kk) * OO + n0;
#pragma unroll
        for (int i = 0; i < 16; ++i) {
            int idx = i * 256 + t;
            int k = idx >> 7, n = idx & 127;
            Bst[n * BPAD + k] = (__bf16)wsrc[(size_t)k * OO + n];
        }
        if (kk + TK < HH)
            __builtin_prefetch(wsrc + (size_t)TK * OO + (t >> 3) * OO + (t & 7) * 16, 0, 3);
        wait_async0();
        __syncthreads();
        v16bf bf = frag_b(Bst, lane, wave * 16);
#pragma unroll
        for (int s = 0; s < 4; ++s) {
            v16bf af = frag_a(Ast + s * 16 * BPAD, lane);
            acc[s] = __builtin_amdgcn_wmma_f32_16x16x32_bf16(false, af, false, bf,
                                                             (short)0, acc[s], false, false);
        }
        __syncthreads();
    }
    // epilogue: bias + routing weight -> LDS -> coalesced float4 stores
    int cn = wave * 16 + (lane & 15);
    float bias = b2[(size_t)e * OO + n0 + cn];
#pragma unroll
    for (int s = 0; s < 4; ++s)
#pragma unroll
        for (int r = 0; r < 8; ++r) {
            int m = s * 16 + r + 8 * (lane >> 4);
            Yst[m][cn] = (acc[s][r] + bias) * wtl[m];
        }
    __syncthreads();
#pragma unroll
    for (int i = 0; i < 8; ++i) {
        int idx = i * 256 + t;
        int m = idx >> 5, n4 = (idx & 31) * 4;
        if (m < valid)
            *(float4*)&yslot[(size_t)dstl[m] * OO + n0 + n4] =
                *(const float4*)&Yst[m][n4];
    }
}

// ---------------------------------------------------------------------------
// Kernel 6: combine the K=2 weighted expert outputs per token
// ---------------------------------------------------------------------------
__global__ __launch_bounds__(256) void moe_reduce(
    const float* __restrict__ yslot, float* __restrict__ out) {
    int i = blockIdx.x * 256 + threadIdx.x;   // i = b*OO + o
    int b = i >> 10, o = i & 1023;
    out[i] = yslot[(size_t)(b * 2) * OO + o] + yslot[(size_t)(b * 2 + 1) * OO + o];
}

// ---------------------------------------------------------------------------
extern "C" void kernel_launch(void* const* d_in, const int* in_sizes, int n_in,
                              void* d_out, int out_size, void* d_ws, size_t ws_size,
                              hipStream_t stream) {
    const float* x      = (const float*)d_in[0];
    const float* gate_w = (const float*)d_in[1];
    const float* gate_b = (const float*)d_in[2];
    const float* w1     = (const float*)d_in[3];
    const float* b1     = (const float*)d_in[4];
    const float* w2     = (const float*)d_in[5];
    const float* b2     = (const float*)d_in[6];
    float* out = (float*)d_out;

    // workspace carve-up
    const int ROWS = BB * KK;          // total grouped rows
    const int ROWS_PAD = ROWS + TM;    // slack for tile over-read
    char* w = (char*)d_ws;
    int*    counts = (int*)w;                       // [8]
    int*    fill   = counts + 8;                    // [8]
    int*    offs   = counts + 16;                   // [8]
    int2*   tkidx  = (int2*)(w + 128);
    float2* tkw    = (float2*)(w + 128 + (size_t)BB * 8);
    int*    rtok   = (int*)(w + 128 + (size_t)BB * 16);
    int*    rdst   = rtok + ROWS_PAD;
    float*  rwt    = (float*)(rdst + ROWS_PAD);
    size_t  hoff   = 128 + (size_t)BB * 16 + 3u * ROWS_PAD * 4;
    hoff = (hoff + 255) & ~(size_t)255;
    __bf16* hbuf   = (__bf16*)(w + hoff);                         // [ROWS_PAD][HH] bf16
    size_t  yoff   = hoff + (size_t)ROWS_PAD * HH * 2;
    yoff = (yoff + 255) & ~(size_t)255;
    float*  yslot  = (float*)(w + yoff);                          // [ROWS][OO] f32

    moe_zero16 <<<1, 16, 0, stream>>>(counts);
    moe_router <<<BB / 8, 256, 0, stream>>>(x, gate_w, gate_b, counts, tkidx, tkw);
    moe_scan   <<<1, 32, 0, stream>>>(counts, offs);
    moe_scatter<<<BB / 256, 256, 0, stream>>>(tkidx, tkw, offs, fill, rtok, rdst, rwt);
    moe_h_gemm <<<dim3(HH / TN, BB / TM, EE), 256, 0, stream>>>(
        x, w1, b1, counts, offs, rtok, hbuf);
    moe_y_gemm <<<dim3(OO / TN, BB / TM, EE), 256, 0, stream>>>(
        hbuf, w2, b2, counts, offs, rdst, rwt, yslot);
    moe_reduce <<<(BB * OO) / 256, 256, 0, stream>>>(yslot, out);
}
